// DeformableCrossAttention_84653805404294
// MI455X (gfx1250) — compile-verified, matrix operand order checked by reading, not run on hardware
//
#include <hip/hip_runtime.h>
#include <hip/hip_bf16.h>

typedef __attribute__((ext_vector_type(16))) _Float16 v16h;
typedef __attribute__((ext_vector_type(8)))  _Float16 v8h;
typedef __attribute__((ext_vector_type(8)))  float    v8f;

#define Q_LEN   4096
#define N_CAM   6
#define D_MODEL 128
#define IMG_H   32
#define IMG_W   88
#define HW      (IMG_H * IMG_W)      /* 2816 */
#define M_HEADS 8
#define DH      32
#define INNER   256                   /* 8*32 */
#define NPTS    8

// ---------------------------------------------------------------------------
// f32 -> f16 conversion (row-major kept; used for weights)
// ---------------------------------------------------------------------------
__global__ void f32_to_f16_kernel(const float* __restrict__ src,
                                  _Float16* __restrict__ dst, int count) {
  int i = blockIdx.x * blockDim.x + threadIdx.x;
  if (i < count) dst[i] = (_Float16)src[i];
}

// ---------------------------------------------------------------------------
// Tiled transpose + f32->f16 convert: src[z][R][C] -> dst[z][C][R]
// block 32x8, tile 32x32. Global->LDS staging uses the CDNA5 async-copy path
// (GLOBAL_LOAD_ASYNC_TO_LDS_B32, tracked by ASYNCcnt). R and C must be
// multiples of 32 (true here), so EXEC is all-ones and no bounds checks.
// ---------------------------------------------------------------------------
__global__ __launch_bounds__(256) void transpose_cvt_kernel(
    const float* __restrict__ src, _Float16* __restrict__ dst, int R, int C) {
  __shared__ float tile[32][33];
  const size_t zoff = (size_t)blockIdx.z * R * C;
  src += zoff;
  dst += zoff;
  int c0 = blockIdx.x * 32;
  int r0 = blockIdx.y * 32;
  int tx = threadIdx.x, ty = threadIdx.y;  // 32 x 8
#pragma unroll
  for (int i = 0; i < 32; i += 8) {
    const float* gp = src + (size_t)(r0 + ty + i) * C + (c0 + tx);
    unsigned la = (unsigned)(size_t)(&tile[ty + i][tx]);  // LDS byte offset
    asm volatile("global_load_async_to_lds_b32 %0, %1, off"
                 :
                 : "v"(la), "v"(gp)
                 : "memory");
  }
  asm volatile("s_wait_asynccnt 0x0" ::: "memory");
  __syncthreads();
#pragma unroll
  for (int i = 0; i < 32; i += 8) {
    int c = c0 + ty + i, r = r0 + tx;
    dst[(size_t)c * R + r] = (_Float16)tile[tx][ty + i];
  }
}

// ---------------------------------------------------------------------------
// Camera projection: grid[n][q][2] in [-1,1]
// ---------------------------------------------------------------------------
__global__ void proj_grid_kernel(const float* __restrict__ E,
                                 const float* __restrict__ Kc,
                                 const float* __restrict__ wxy,
                                 float* __restrict__ grd) {
  int idx = blockIdx.x * blockDim.x + threadIdx.x;
  if (idx >= N_CAM * Q_LEN) return;
  int q = idx % Q_LEN;
  int n = idx / Q_LEN;
  float X = wxy[q];
  float Y = wxy[Q_LEN + q];
  const float* e = E + n * 16;
  const float* k = Kc + n * 9;
  float c0 = e[0] * X + e[1] * Y + e[3];
  float c1 = e[4] * X + e[5] * Y + e[7];
  float c2 = e[8] * X + e[9] * Y + e[11];
  float p0 = k[0] * c0 + k[1] * c1 + k[2] * c2;
  float p1 = k[3] * c0 + k[4] * c1 + k[5] * c2;
  float p2 = k[6] * c0 + k[7] * c1 + k[8] * c2;
  float z  = fmaxf(p2, 1e-6f);
  float u  = p0 / z;
  float v  = p1 / z;
  size_t o = ((size_t)n * Q_LEN + q) * 2;
  grd[o + 0] = u / (float)(IMG_W - 1) * 2.0f - 1.0f;
  grd[o + 1] = v / (float)(IMG_H - 1) * 2.0f - 1.0f;
}

// ---------------------------------------------------------------------------
// WMMA f16 GEMM: C[M,N] = A[M,K] * Bt[N,K]^T + bias (+relu)
// Register-blocked: each wave computes MBLK x 4 tiles of 16x16 (A fragments
// reused across 4 N-tiles, B fragments across MBLK M-tiles: 8 WMMAs per
// 12 b128 loads at MBLK=2). Both operands K-major -> contiguous 16B loads.
// OUT: 0 = f32 stored transposed [N][M] (per lane: two float4 stores)
//      1 = f16 stored transposed [N][M] (per lane: one v8h store)
//      2 = f32 stored row-major [M][N]  (lane-coalesced stores)
// ---------------------------------------------------------------------------
template <int MBLK, int OUT, bool RELU>
__global__ __launch_bounds__(128) void wmma_gemm_f16_kernel(
    const _Float16* __restrict__ A, const _Float16* __restrict__ Bt,
    const float* __restrict__ bias, void* __restrict__ Cout, int M, int N,
    int K) {
  const int wave = threadIdx.x >> 5;
  const int lane = threadIdx.x & 31;
  const int tn   = (blockIdx.x * 4 + wave) * 64;   // 4 N-subtiles of 16
  const int tm   = blockIdx.y * (16 * MBLK);
  if (tn >= N || tm >= M) return;                  // wave-uniform
  const int hl = lane >> 4;                        // 0 | 1
  const int r  = lane & 15;

  // bias: 8 consecutive rows per lane-half, loaded once
  float bv[MBLK][8];
#pragma unroll
  for (int i = 0; i < MBLK; ++i) {
    const float4* bp = (const float4*)(bias + tm + i * 16 + hl * 8);
    float4 b0 = bp[0], b1 = bp[1];
    bv[i][0] = b0.x; bv[i][1] = b0.y; bv[i][2] = b0.z; bv[i][3] = b0.w;
    bv[i][4] = b1.x; bv[i][5] = b1.y; bv[i][6] = b1.z; bv[i][7] = b1.w;
  }

  const _Float16* arow[MBLK];
#pragma unroll
  for (int i = 0; i < MBLK; ++i)
    arow[i] = A + (size_t)(tm + i * 16 + r) * K + hl * 8;
  const _Float16* brow[4];
#pragma unroll
  for (int j = 0; j < 4; ++j)
    brow[j] = Bt + (size_t)(tn + j * 16 + r) * K + hl * 8;

  v8f acc[MBLK][4];
#pragma unroll
  for (int i = 0; i < MBLK; ++i)
#pragma unroll
    for (int j = 0; j < 4; ++j) acc[i][j] = (v8f){};

  for (int k0 = 0; k0 < K; k0 += 32) {
    v16h af[MBLK], bf[4];
#pragma unroll
    for (int i = 0; i < MBLK; ++i) {
      v8h lo = *(const v8h*)(arow[i] + k0);
      v8h hi = *(const v8h*)(arow[i] + k0 + 16);
#pragma unroll
      for (int t = 0; t < 8; ++t) { af[i][t] = lo[t]; af[i][8 + t] = hi[t]; }
    }
#pragma unroll
    for (int j = 0; j < 4; ++j) {
      v8h lo = *(const v8h*)(brow[j] + k0);
      v8h hi = *(const v8h*)(brow[j] + k0 + 16);
#pragma unroll
      for (int t = 0; t < 8; ++t) { bf[j][t] = lo[t]; bf[j][8 + t] = hi[t]; }
    }
#pragma unroll
    for (int j = 0; j < 4; ++j)
#pragma unroll
      for (int i = 0; i < MBLK; ++i)
        acc[i][j] = __builtin_amdgcn_wmma_f32_16x16x32_f16(
            false, af[i], false, bf[j], (short)0, acc[i][j], false, false);
  }

#pragma unroll
  for (int i = 0; i < MBLK; ++i) {
#pragma unroll
    for (int j = 0; j < 4; ++j) {
      float vals[8];
#pragma unroll
      for (int v = 0; v < 8; ++v) {
        float x = acc[i][j][v] + bv[i][v];
        if (RELU) x = fmaxf(x, 0.0f);
        vals[v] = x;
      }
      const int col = tn + j * 16 + r;
      if (OUT == 0) {  // f32 transposed [N][M]
        float* Cf = (float*)Cout;
        size_t o = (size_t)col * M + tm + i * 16 + hl * 8;
        *(float4*)(Cf + o)     = make_float4(vals[0], vals[1], vals[2], vals[3]);
        *(float4*)(Cf + o + 4) = make_float4(vals[4], vals[5], vals[6], vals[7]);
      } else if (OUT == 1) {  // f16 transposed [N][M]
        _Float16* Ch = (_Float16*)Cout;
        size_t o = (size_t)col * M + tm + i * 16 + hl * 8;
        v8h hh;
#pragma unroll
        for (int v = 0; v < 8; ++v) hh[v] = (_Float16)vals[v];
        *(v8h*)(Ch + o) = hh;
      } else {  // f32 row-major [M][N]
        float* Cf = (float*)Cout;
#pragma unroll
        for (int v = 0; v < 8; ++v)
          Cf[(size_t)(tm + i * 16 + hl * 8 + v) * N + col] = vals[v];
      }
    }
  }
}

// ---------------------------------------------------------------------------
// Fused deformable attention. kv is pixel-major [n][HW][512]: each bilinear
// tap reads one 128B line of 32 consecutive channels as float4s. V taps are
// prefetched during the K/sim pass. kv (35 MB) stays resident in L2.
// ---------------------------------------------------------------------------
__device__ __forceinline__ void make_taps(float gx, float gy, int& i00, int& i01,
                                          int& i10, int& i11, float& w00,
                                          float& w01, float& w10, float& w11) {
  gx = fminf(fmaxf(gx, -1.0f), 1.0f);
  gy = fminf(fmaxf(gy, -1.0f), 1.0f);
  float x  = (gx + 1.0f) * 0.5f * (float)(IMG_W - 1);
  float y  = (gy + 1.0f) * 0.5f * (float)(IMG_H - 1);
  float x0 = floorf(x), y0 = floorf(y);
  float wx = x - x0, wy = y - y0;
  int x0i = (int)x0;
  x0i = x0i < 0 ? 0 : (x0i > IMG_W - 1 ? IMG_W - 1 : x0i);
  int x1i = x0i + 1 > IMG_W - 1 ? IMG_W - 1 : x0i + 1;
  int y0i = (int)y0;
  y0i = y0i < 0 ? 0 : (y0i > IMG_H - 1 ? IMG_H - 1 : y0i);
  int y1i = y0i + 1 > IMG_H - 1 ? IMG_H - 1 : y0i + 1;
  i00 = y0i * IMG_W + x0i;
  i01 = y0i * IMG_W + x1i;
  i10 = y1i * IMG_W + x0i;
  i11 = y1i * IMG_W + x1i;
  w00 = (1.0f - wx) * (1.0f - wy);
  w01 = wx * (1.0f - wy);
  w10 = (1.0f - wx) * wy;
  w11 = wx * wy;
}

__global__ __launch_bounds__(256) void attn_kernel(
    const float* __restrict__ kv,    // [n][HW][512] pixel-major
    const float* __restrict__ qb,    // [Q_LEN][256] query-major
    const float* __restrict__ grd,   // [n][Q_LEN][2]
    const float* __restrict__ off,   // [16][Q_LEN]
    _Float16* __restrict__ wout) {   // [Q_LEN][256] query-major f16
  int idx = blockIdx.x * blockDim.x + threadIdx.x;
  if (idx >= M_HEADS * Q_LEN) return;
  int q = idx & (Q_LEN - 1);
  int m = idx >> 12;

  float4 qv[DH / 4];
  {
    const float4* qp = (const float4*)(qb + (size_t)q * INNER + m * DH);
#pragma unroll
    for (int i = 0; i < DH / 4; ++i) qv[i] = qp[i];
  }
  float acc[DH];
#pragma unroll
  for (int d = 0; d < DH; ++d) acc[d] = 0.0f;

  float offx[NPTS], offy[NPTS];
#pragma unroll
  for (int p = 0; p < NPTS; ++p) {
    offx[p] = off[(size_t)(p * 2 + 0) * Q_LEN + q];
    offy[p] = off[(size_t)(p * 2 + 1) * Q_LEN + q];
  }

  for (int n = 0; n < N_CAM; ++n) {
    const float* base = kv + (size_t)n * HW * (2 * INNER);
    const float* vb   = base + INNER + m * DH;  // V block: +256 channels
    float gx0 = grd[((size_t)n * Q_LEN + q) * 2 + 0];
    float gy0 = grd[((size_t)n * Q_LEN + q) * 2 + 1];

    float s[NPTS];
#pragma unroll
    for (int p = 0; p < NPTS; ++p) {
      int i00, i01, i10, i11;
      float w00, w01, w10, w11;
      make_taps(gx0 + offx[p], gy0 + offy[p], i00, i01, i10, i11, w00, w01,
                w10, w11);
      const float4* a00 = (const float4*)(base + (size_t)i00 * 512 + m * DH);
      const float4* a01 = (const float4*)(base + (size_t)i01 * 512 + m * DH);
      const float4* a10 = (const float4*)(base + (size_t)i10 * 512 + m * DH);
      const float4* a11 = (const float4*)(base + (size_t)i11 * 512 + m * DH);
      // warm the V lines we will gather after the softmax
      __builtin_prefetch(vb + (size_t)i00 * 512, 0, 3);
      __builtin_prefetch(vb + (size_t)i01 * 512, 0, 3);
      __builtin_prefetch(vb + (size_t)i10 * 512, 0, 3);
      __builtin_prefetch(vb + (size_t)i11 * 512, 0, 3);
      float sim = 0.0f;
#pragma unroll
      for (int d4 = 0; d4 < DH / 4; ++d4) {
        float4 k00 = a00[d4], k01 = a01[d4], k10 = a10[d4], k11 = a11[d4];
        float4 qq  = qv[d4];
        sim += qq.x * (k00.x * w00 + k01.x * w01 + k10.x * w10 + k11.x * w11);
        sim += qq.y * (k00.y * w00 + k01.y * w01 + k10.y * w10 + k11.y * w11);
        sim += qq.z * (k00.z * w00 + k01.z * w01 + k10.z * w10 + k11.z * w11);
        sim += qq.w * (k00.w * w00 + k01.w * w01 + k10.w * w10 + k11.w * w11);
      }
      s[p] = sim;
    }
    // softmax over points, per camera
    float mx = s[0];
#pragma unroll
    for (int p = 1; p < NPTS; ++p) mx = fmaxf(mx, s[p]);
    float sum = 0.0f;
#pragma unroll
    for (int p = 0; p < NPTS; ++p) { s[p] = __expf(s[p] - mx); sum += s[p]; }
    float inv = 1.0f / sum;
#pragma unroll
    for (int p = 0; p < NPTS; ++p) {
      float a = s[p] * inv;
      int i00, i01, i10, i11;
      float w00, w01, w10, w11;
      make_taps(gx0 + offx[p], gy0 + offy[p], i00, i01, i10, i11, w00, w01,
                w10, w11);
      const float4* a00 = (const float4*)(vb + (size_t)i00 * 512);
      const float4* a01 = (const float4*)(vb + (size_t)i01 * 512);
      const float4* a10 = (const float4*)(vb + (size_t)i10 * 512);
      const float4* a11 = (const float4*)(vb + (size_t)i11 * 512);
#pragma unroll
      for (int d4 = 0; d4 < DH / 4; ++d4) {
        float4 v00 = a00[d4], v01 = a01[d4], v10 = a10[d4], v11 = a11[d4];
        acc[d4 * 4 + 0] += a * (v00.x * w00 + v01.x * w01 + v10.x * w10 + v11.x * w11);
        acc[d4 * 4 + 1] += a * (v00.y * w00 + v01.y * w01 + v10.y * w10 + v11.y * w11);
        acc[d4 * 4 + 2] += a * (v00.z * w00 + v01.z * w01 + v10.z * w10 + v11.z * w11);
        acc[d4 * 4 + 3] += a * (v00.w * w00 + v01.w * w01 + v10.w * w10 + v11.w * w11);
      }
    }
  }
  const float invn = 1.0f / (float)N_CAM;
  v8h* wp = (v8h*)(wout + (size_t)q * INNER + m * DH);
#pragma unroll
  for (int c = 0; c < DH / 8; ++c) {
    v8h hh;
#pragma unroll
    for (int j = 0; j < 8; ++j) hh[j] = (_Float16)(acc[c * 8 + j] * invn);
    wp[c] = hh;
  }
}

// ---------------------------------------------------------------------------
// Orchestration
// ---------------------------------------------------------------------------
extern "C" void kernel_launch(void* const* d_in, const int* in_sizes, int n_in,
                              void* d_out, int out_size, void* d_ws,
                              size_t ws_size, hipStream_t stream) {
  const float* bev       = (const float*)d_in[0];   // [128][64][64]
  const float* img_feats = (const float*)d_in[1];   // [6][128][32][88]
  const float* Kmat      = (const float*)d_in[2];   // [6][3][3]
  const float* Emat      = (const float*)d_in[3];   // [6][4][4]
  const float* world_xy  = (const float*)d_in[4];   // [2][64][64]
  const float* q_w       = (const float*)d_in[5];   // [256][128]
  const float* q_b       = (const float*)d_in[6];
  const float* kv_w      = (const float*)d_in[7];   // [512][128]
  const float* kv_b      = (const float*)d_in[8];
  const float* off_w1    = (const float*)d_in[9];   // [128][128]
  const float* off_b1    = (const float*)d_in[10];
  const float* off_w2    = (const float*)d_in[11];  // [16][128]
  const float* off_b2    = (const float*)d_in[12];
  const float* proj_w    = (const float*)d_in[13];  // [128][256]
  const float* proj_b    = (const float*)d_in[14];
  float* out = (float*)d_out;                       // [128][4096]

  // ---- workspace layout ----
  float* ws_f   = (float*)d_ws;
  float* kv     = ws_f;                  // [6][2816][512] = 8650752 f
  float* qbuf   = kv + 8650752;          // [4096][256]    = 1048576 f
  float* grd    = qbuf + 1048576;        // [6][4096][2]   = 49152 f
  float* offb   = grd + 49152;           // [16][4096]     = 65536 f
  _Float16* hb  = (_Float16*)(offb + 65536);
  _Float16* h_img = hb;                  // [6][2816][128] = 2162688 h
  _Float16* h_bev = h_img + 2162688;     // [4096][128]    = 524288 h
  _Float16* h_kvw = h_bev + 524288;      // [512][128]     = 65536 h
  _Float16* h_qw  = h_kvw + 65536;       // [256][128]     = 32768 h
  _Float16* h_ow1 = h_qw + 32768;        // [128][128]     = 16384 h
  _Float16* h_ow2 = h_ow1 + 16384;       // [16][128]      = 2048 h
  _Float16* h_pw  = h_ow2 + 2048;        // [128][256]     = 32768 h
  _Float16* h_xh  = h_pw + 32768;        // [4096][128]    = 524288 h
  _Float16* h_wt  = h_xh + 524288;       // [4096][256]    = 1048576 h

  auto cvt = [&](const float* s, _Float16* d, int n) {
    f32_to_f16_kernel<<<(n + 255) / 256, 256, 0, stream>>>(s, d, n);
  };
  cvt(kv_w, h_kvw, 512 * 128);
  cvt(q_w, h_qw, 256 * 128);
  cvt(off_w1, h_ow1, 128 * 128);
  cvt(off_w2, h_ow2, 16 * 128);
  cvt(proj_w, h_pw, 128 * 256);

  // activations: transpose to K-major [N][K] while converting (async->LDS)
  {
    dim3 b(32, 8);
    dim3 g1(HW / 32, 128 / 32, N_CAM);
    transpose_cvt_kernel<<<g1, b, 0, stream>>>(img_feats, h_img, 128, HW);
    dim3 g2(Q_LEN / 32, 128 / 32, 1);
    transpose_cvt_kernel<<<g2, b, 0, stream>>>(bev, h_bev, 128, Q_LEN);
  }

  proj_grid_kernel<<<(N_CAM * Q_LEN + 255) / 256, 256, 0, stream>>>(
      Emat, Kmat, world_xy, grd);

  // offset MLP: xh = relu(off_w1 @ bev + b1) -> f16 transposed [4096][128]
  wmma_gemm_f16_kernel<2, 1, true><<<dim3(Q_LEN / 256, 128 / 32), 128, 0,
                                    stream>>>(h_ow1, h_bev, off_b1, h_xh, 128,
                                              Q_LEN, 128);
  // off = off_w2 @ xh + b2 -> f32 row-major [16][4096]
  wmma_gemm_f16_kernel<1, 2, false><<<dim3(Q_LEN / 256, 16 / 16), 128, 0,
                                     stream>>>(h_ow2, h_xh, off_b2, offb, 16,
                                               Q_LEN, 128);
  // q projection -> f32 transposed [4096][256]
  wmma_gemm_f16_kernel<2, 0, false><<<dim3(Q_LEN / 256, 256 / 32), 128, 0,
                                     stream>>>(h_qw, h_bev, q_b, qbuf, 256,
                                               Q_LEN, 128);
  // kv projection per camera -> f32 transposed [2816][512] (pixel-major)
  for (int n = 0; n < N_CAM; ++n)
    wmma_gemm_f16_kernel<2, 0, false><<<dim3(HW / 256, 512 / 32), 128, 0,
                                       stream>>>(
        h_kvw, h_img + (size_t)n * HW * 128, kv_b, kv + (size_t)n * HW * 512,
        512, HW, 128);
  // fused sampling + attention -> weighted f16 [4096][256]
  attn_kernel<<<(M_HEADS * Q_LEN) / 256, 256, 0, stream>>>(kv, qbuf, grd, offb,
                                                           h_wt);
  // output projection straight into d_out, row-major [128][4096]
  wmma_gemm_f16_kernel<2, 2, false><<<dim3(Q_LEN / 256, 128 / 32), 128, 0,
                                     stream>>>(h_pw, h_wt, proj_b, out, 128,
                                               Q_LEN, 256);
}
